// SinkhornNet_11510512353773
// MI455X (gfx1250) — compile-verified, matrix operand order checked by reading, not run on hardware
//
#include <hip/hip_runtime.h>
#include <hip/hip_bf16.h>

// ---------------------------------------------------------------------------
// SinkhornNet forward for MI455X (gfx1250, wave32).
// Conv layers: implicit GEMM via v_wmma_f32_16x16x32_f16 (f16 in, f32 acc).
// conv2/3/4: 32x32 wave tiles (4 WMMAs/K-step), A/B tiles staged in LDS with
// double-buffered GLOBAL_LOAD_ASYNC_TO_LDS_B32 (ASYNCcnt-tracked async path).
// conv1 (K=75, needs zero-pad guards) uses a batched synchronous path.
// FC layers: per-wave WMMA GEMM. Sinkhorn: register-resident 6x6 logsumexp.
// ---------------------------------------------------------------------------

typedef __attribute__((ext_vector_type(16))) _Float16 v16h_t;
typedef __attribute__((ext_vector_type(8)))  float    v8f_t;

// Build a 16x32 f16 A/B fragment from an f32 LDS row (32 consecutive floats).
// CDNA5 16-bit A-matrix 16x32 layout:
//   lanes 0-15 : K 0..7 in v0-3, K 16..23 in v4-7
//   lanes 16-31: K 8..15        , K 24..31
__device__ __forceinline__ v16h_t frag_from_f32(const float* __restrict__ row,
                                                int half) {
    v16h_t f;
    const int b = half * 8;
#pragma unroll
    for (int i = 0; i < 8; ++i) {
        f[i]     = (_Float16)row[b + i];
        f[i + 8] = (_Float16)row[b + 16 + i];
    }
    return f;
}

// Async global->LDS dword copy (gfx1250). lds_off = low 32 bits of the
// generic pointer to shared memory (== wave-relative LDS byte address).
__device__ __forceinline__ void async_copy_b32(unsigned lds_off,
                                               const float* gptr) {
    asm volatile("global_load_async_to_lds_b32 %0, %1, off"
                 :: "v"(lds_off), "v"((unsigned long long)(uintptr_t)gptr)
                 : "memory");
}

template <int N>
__device__ __forceinline__ void wait_asynccnt_le() {
    asm volatile("s_wait_asynccnt %0" :: "i"(N) : "memory");
}

// ---------------------------------------------------------------------------
// Implicit-GEMM 5x5 VALID conv + bias + ReLU, NCHW f32 in/out.
//   M = N*Hout*Wout, N = Cout, K = Cin*25.
// Wave tile = (16*MR) x 32 -> MR A-fragments, 2 B-fragments, 2*MR WMMAs/step.
// ---------------------------------------------------------------------------
template <int Cin, int Cout, int Hin, int Win, int M_WAVES, int N_WAVES, int MR>
__global__ __launch_bounds__(32 * M_WAVES * N_WAVES)
void conv5x5_relu_wmma(const float* __restrict__ im,
                       const float* __restrict__ w,
                       const float* __restrict__ bias,
                       float* __restrict__ out) {
    constexpr int Hout = Hin - 4, Wout = Win - 4;
    constexpr int KT = Cin * 25;
    constexpr int M_TILE = 16 * MR * M_WAVES;
    constexpr int N_TILE = 32 * N_WAVES;
    constexpr int NTHREADS = 32 * M_WAVES * N_WAVES;
    constexpr int KSTEPS = (KT + 31) / 32;
    constexpr bool ASYNC = (KT % 32) == 0;   // no zero-pad guards needed
    constexpr int NBUF = ASYNC ? 2 : 1;
    constexpr int A_PER = (M_TILE * 32) / NTHREADS;
    constexpr int B_PER = (N_TILE * 32) / NTHREADS;
    constexpr int PER_STEP = A_PER + B_PER;  // async issues per thread per step

    __shared__ alignas(16) float Asf[NBUF][M_TILE * 32];
    __shared__ alignas(16) float Bsf[NBUF][N_TILE * 32];

    const int t    = threadIdx.x;
    const int wave = t >> 5;
    const int lane = t & 31;
    const int wm   = wave % M_WAVES;
    const int wn   = wave / M_WAVES;
    const int lrow = lane & 15;
    const int half = lane >> 4;
    const int m0   = blockIdx.x * M_TILE;
    const int n0   = blockIdx.y * N_TILE;

    // im2col source address for A-tile element e of K-step base k0.
    auto a_gaddr = [&](int e, int k0) -> const float* {
        const int m  = e % M_TILE;   // consecutive threads -> consecutive m
        const int kk = k0 + e / M_TILE;
        const int mg = m0 + m;
        const int ox = mg % Wout;
        const int r1 = mg / Wout;
        const int oy = r1 % Hout;
        const int n  = r1 / Hout;
        const int ci = kk / 25;
        const int rr = kk % 25;
        const int ky = rr / 5, kx = rr % 5;
        return &im[(((long)n * Cin + ci) * Hin + (oy + ky)) * Win + (ox + kx)];
    };

    v8f_t acc0[MR], acc1[MR];
#pragma unroll
    for (int r = 0; r < MR; ++r) { acc0[r] = (v8f_t){}; acc1[r] = (v8f_t){}; }

    auto compute_step = [&](const float* ab, const float* bb) {
        v16h_t bf0 = frag_from_f32(&bb[(wn * 32 + lrow) * 32], half);
        v16h_t bf1 = frag_from_f32(&bb[(wn * 32 + 16 + lrow) * 32], half);
#pragma unroll
        for (int r = 0; r < MR; ++r) {
            v16h_t af =
                frag_from_f32(&ab[((wm * MR + r) * 16 + lrow) * 32], half);
            acc0[r] = __builtin_amdgcn_wmma_f32_16x16x32_f16(
                false, af, false, bf0, (short)0, acc0[r], false, false);
            acc1[r] = __builtin_amdgcn_wmma_f32_16x16x32_f16(
                false, af, false, bf1, (short)0, acc1[r], false, false);
        }
    };

    if constexpr (ASYNC) {
        auto issue_batch = [&](int ks) {
            const int k0 = ks * 32;
            float* abuf = Asf[ks & 1];
            float* bbuf = Bsf[ks & 1];
#pragma unroll
            for (int it = 0; it < A_PER; ++it) {
                const int e = t + it * NTHREADS;
                const int m = e % M_TILE, k = e / M_TILE;
                async_copy_b32((unsigned)(uintptr_t)&abuf[m * 32 + k],
                               a_gaddr(e, k0));
            }
#pragma unroll
            for (int it = 0; it < B_PER; ++it) {
                const int e = t + it * NTHREADS;
                const int k = e & 31, nn = e >> 5;  // k fastest: coalesced
                async_copy_b32((unsigned)(uintptr_t)&bbuf[nn * 32 + k],
                               &w[(long)(n0 + nn) * KT + k0 + k]);
            }
        };

        issue_batch(0);
        for (int ks = 0; ks < KSTEPS; ++ks) {
            if (ks + 1 < KSTEPS) {
                issue_batch(ks + 1);                 // prefetch next buffer
                wait_asynccnt_le<PER_STEP>();        // drain batch(ks) only
            } else {
                wait_asynccnt_le<0>();
            }
            __syncthreads();
            compute_step(Asf[ks & 1], Bsf[ks & 1]);
            __syncthreads();
        }
    } else {
        // Guarded path (conv1, K=75): batch global loads into registers so
        // they overlap, then store to LDS as f32.
        for (int ks = 0; ks < KSTEPS; ++ks) {
            const int k0 = ks * 32;
            float va[A_PER];
#pragma unroll
            for (int it = 0; it < A_PER; ++it) {
                const int e  = t + it * NTHREADS;
                const int kk = k0 + e / M_TILE;
                va[it] = (kk < KT) ? *a_gaddr(e, k0) : 0.0f;
            }
#pragma unroll
            for (int it = 0; it < A_PER; ++it) {
                const int e = t + it * NTHREADS;
                Asf[0][(e % M_TILE) * 32 + e / M_TILE] = va[it];
            }
            float vb[B_PER];
#pragma unroll
            for (int it = 0; it < B_PER; ++it) {
                const int e  = t + it * NTHREADS;
                const int kk = k0 + (e & 31);
                vb[it] = (kk < KT) ? w[(long)(n0 + (e >> 5)) * KT + kk] : 0.0f;
            }
#pragma unroll
            for (int it = 0; it < B_PER; ++it) {
                const int e = t + it * NTHREADS;
                Bsf[0][(e >> 5) * 32 + (e & 31)] = vb[it];
            }
            __syncthreads();
            compute_step(Asf[0], Bsf[0]);
            __syncthreads();
        }
    }

    // C/D layout: lane -> column, VGPR i -> row (half selects +8). Bias+ReLU.
    const int col0 = n0 + wn * 32 + lrow;
    const int col1 = col0 + 16;
    const float b0 = bias[col0];
    const float b1 = bias[col1];
#pragma unroll
    for (int r = 0; r < MR; ++r) {
#pragma unroll
        for (int i = 0; i < 8; ++i) {
            const int mg = m0 + (wm * MR + r) * 16 + half * 8 + i;
            const int ox = mg % Wout;
            const int r1 = mg / Wout;
            const int oy = r1 % Hout;
            const int n  = r1 / Hout;
            float v0 = acc0[r][i] + b0; v0 = v0 > 0.0f ? v0 : 0.0f;
            float v1 = acc1[r][i] + b1; v1 = v1 > 0.0f ? v1 : 0.0f;
            out[(((long)n * Cout + col0) * Hout + oy) * Wout + ox] = v0;
            out[(((long)n * Cout + col1) * Hout + oy) * Wout + ox] = v1;
        }
    }
}

// ---------------------------------------------------------------------------
// Small GEMM: out[M x N] = relu?(A[M x KT] * W[N x KT]^T + bias).
// One wave per 16x16 tile; fragments gathered from global with K/N guards.
// ---------------------------------------------------------------------------
template <int KT, bool RELU>
__global__ __launch_bounds__(32)
void gemm_bias_wmma(const float* __restrict__ A,
                    const float* __restrict__ W,
                    const float* __restrict__ bias,
                    float* __restrict__ out, int N) {
    const int lane = threadIdx.x;
    const int lrow = lane & 15, half = lane >> 4;
    const int m0 = blockIdx.x * 16, n0 = blockIdx.y * 16;
    const int row = m0 + lrow;
    const int col = n0 + lrow;
    v8f_t acc = {};
    for (int k0 = 0; k0 < KT; k0 += 32) {
        v16h_t af, bf;
#pragma unroll
        for (int i = 0; i < 16; ++i) {
            const int ka = k0 + half * 8 + (i < 8 ? i : i + 8);
            af[i] = (ka < KT) ? (_Float16)A[(long)row * KT + ka] : (_Float16)0.0f;
            bf[i] = (ka < KT && col < N) ? (_Float16)W[(long)col * KT + ka]
                                         : (_Float16)0.0f;
        }
        acc = __builtin_amdgcn_wmma_f32_16x16x32_f16(false, af, false, bf,
                                                     (short)0, acc, false, false);
    }
    if (col < N) {
        const float b = bias[col];
#pragma unroll
        for (int i = 0; i < 8; ++i) {
            const int m = m0 + half * 8 + i;
            float v = acc[i] + b;
            if (RELU) v = v > 0.0f ? v : 0.0f;
            out[(long)m * N + col] = v;
        }
    }
}

// ---------------------------------------------------------------------------
// 2x2/2 max-pool, NCHW.
// ---------------------------------------------------------------------------
__global__ void maxpool2x2(const float* __restrict__ in, float* __restrict__ out,
                           int H, int W, long total) {
    long i = (long)blockIdx.x * blockDim.x + threadIdx.x;
    if (i >= total) return;
    const int Wo = W >> 1, Ho = H >> 1;
    const int x = (int)(i % Wo);
    long r = i / Wo;
    const int y = (int)(r % Ho);
    r /= Ho;  // r = n*C + c
    const float* p = in + (r * H + 2 * y) * W + 2 * x;
    out[i] = fmaxf(fmaxf(p[0], p[1]), fmaxf(p[W], p[W + 1]));
}

// ---------------------------------------------------------------------------
// Gumbel-Sinkhorn on 1280 6x6 matrices, 20 iterations of stable row/col
// logsumexp, then ordered = sink^T @ seq. One thread per matrix.
// ---------------------------------------------------------------------------
__global__ __launch_bounds__(256)
void sinkhorn_apply(const float* __restrict__ log_alpha, // [256, 36]
                    const float* __restrict__ gumbel,    // [1280, 36]
                    const float* __restrict__ seq,       // [256, 6]
                    float* __restrict__ out) {           // [1280, 6]
    const int idx = blockIdx.x * 256 + threadIdx.x;
    if (idx >= 1280) return;
    const int b = idx % 256;  // tile(log_alpha, (NS,1,1)): sample-major

    float la[36];
#pragma unroll
    for (int i = 0; i < 36; ++i)
        la[i] = log_alpha[b * 36 + i] + gumbel[idx * 36 + i];  // noise=1, temp=1

#pragma unroll 1
    for (int it = 0; it < 20; ++it) {
#pragma unroll
        for (int i = 0; i < 6; ++i) {  // rows (axis=2)
            float m = la[i * 6];
#pragma unroll
            for (int j = 1; j < 6; ++j) m = fmaxf(m, la[i * 6 + j]);
            float s = 0.0f;
#pragma unroll
            for (int j = 0; j < 6; ++j) s += __expf(la[i * 6 + j] - m);
            const float lse = m + __logf(s);
#pragma unroll
            for (int j = 0; j < 6; ++j) la[i * 6 + j] -= lse;
        }
#pragma unroll
        for (int j = 0; j < 6; ++j) {  // cols (axis=1)
            float m = la[j];
#pragma unroll
            for (int i = 1; i < 6; ++i) m = fmaxf(m, la[i * 6 + j]);
            float s = 0.0f;
#pragma unroll
            for (int i = 0; i < 6; ++i) s += __expf(la[i * 6 + j] - m);
            const float lse = m + __logf(s);
#pragma unroll
            for (int i = 0; i < 6; ++i) la[i * 6 + j] -= lse;
        }
    }

    float sq[6];
#pragma unroll
    for (int j = 0; j < 6; ++j) sq[j] = seq[b * 6 + j];
#pragma unroll
    for (int i = 0; i < 6; ++i) {
        float s = 0.0f;
#pragma unroll
        for (int j = 0; j < 6; ++j) s += __expf(la[j * 6 + i]) * sq[j];  // P = sink^T
        out[idx * 6 + i] = s;
    }
}

// ---------------------------------------------------------------------------
// Orchestration; workspace ping-pongs two regions (peak ~324 MB):
//   region A (offset 0,  29,491,200 f): x1, p2, p3, p4/flat, t1
//   region B (offset X1, 51,380,224 f): x2, x3, x4, latent/log_alpha
// ---------------------------------------------------------------------------
extern "C" void kernel_launch(void* const* d_in, const int* in_sizes, int n_in,
                              void* d_out, int out_size, void* d_ws, size_t ws_size,
                              hipStream_t stream) {
    (void)in_sizes; (void)n_in; (void)out_size; (void)ws_size;

    const float* seq    = (const float*)d_in[0];
    const float* im     = (const float*)d_in[1];
    const float* gumbel = (const float*)d_in[2];
    const float* w1 = (const float*)d_in[3];
    const float* b1 = (const float*)d_in[4];
    const float* w2 = (const float*)d_in[5];
    const float* b2 = (const float*)d_in[6];
    const float* w3 = (const float*)d_in[7];
    const float* b3 = (const float*)d_in[8];
    const float* w4 = (const float*)d_in[9];
    const float* b4 = (const float*)d_in[10];
    const float* fc_w = (const float*)d_in[11];
    const float* fc_b = (const float*)d_in[12];
    const float* s1_w = (const float*)d_in[13];
    const float* s1_b = (const float*)d_in[14];
    const float* s2_w = (const float*)d_in[15];
    const float* s2_b = (const float*)d_in[16];

    float* ws = (float*)d_ws;
    const long X1 = 256L * 32 * 60 * 60;   // 29,491,200
    float* RA = ws;
    float* RB = ws + X1;

    const long P2 = 256L * 64 * 28 * 28;    // 12,845,056
    const long P3 = 256L * 128 * 12 * 12;   //  4,718,592
    const long P4 = 256L * 256 * 4 * 4;     //  1,048,576

    // conv1: [256,3,64,64] -> [256,32,60,60]  (guarded sync path, K=75)
    conv5x5_relu_wmma<3, 32, 64, 64, 8, 1, 1>
        <<<dim3(7200, 1), 256, 0, stream>>>(im, w1, b1, RA);
    // conv2: -> [256,64,56,56]  (async, 32x32 wave tiles, M_TILE=128)
    conv5x5_relu_wmma<32, 64, 60, 60, 4, 2, 2>
        <<<dim3(6272, 1), 256, 0, stream>>>(RA, w2, b2, RB);
    maxpool2x2<<<(unsigned)((P2 + 255) / 256), 256, 0, stream>>>(RB, RA, 56, 56, P2);
    // conv3: -> [256,128,24,24]
    conv5x5_relu_wmma<64, 128, 28, 28, 4, 2, 2>
        <<<dim3(1152, 2), 256, 0, stream>>>(RA, w3, b3, RB);
    maxpool2x2<<<(unsigned)((P3 + 255) / 256), 256, 0, stream>>>(RB, RA, 24, 24, P3);
    // conv4: -> [256,256,8,8]
    conv5x5_relu_wmma<128, 256, 12, 12, 4, 2, 2>
        <<<dim3(128, 4), 256, 0, stream>>>(RA, w4, b4, RB);
    maxpool2x2<<<(unsigned)((P4 + 255) / 256), 256, 0, stream>>>(RB, RA, 8, 8, P4);

    // fc/s1/s2 chain
    gemm_bias_wmma<4096, true><<<dim3(16, 1), 32, 0, stream>>>(RA, fc_w, fc_b, RB, 16);
    gemm_bias_wmma<16, true><<<dim3(16, 1), 32, 0, stream>>>(RB, s1_w, s1_b, RA, 16);
    gemm_bias_wmma<16, false><<<dim3(16, 3), 32, 0, stream>>>(RA, s2_w, s2_b, RB, 36);

    // Gumbel-Sinkhorn + permutation apply -> d_out [1280,6]
    sinkhorn_apply<<<5, 256, 0, stream>>>(RB, gumbel, seq, (float*)d_out);
}
